// APOBECEditEmbedding_62423054680166
// MI455X (gfx1250) — matrix operand order repaired and checked by reading, not compiled, not run on hardware
//
#include <hip/hip_runtime.h>
#include <math.h>

typedef __attribute__((ext_vector_type(16))) _Float16 v16h;
typedef __attribute__((ext_vector_type(8)))  _Float16 v8h;
typedef __attribute__((ext_vector_type(8)))  float    v8f;

#define B_    512
#define S_    512
#define D_    640
#define H_    8
#define DH_   80
#define DHP_  96          // DH zero-padded to multiple of 32
#define DE_   256
#define FUSE_ 768

__device__ __forceinline__ float gelu_exact(float v) {
  return 0.5f * v * (1.0f + erff(v * 0.70710678118654752440f));
}

// ---------------------------------------------------------------------------
// Generic WMMA f16 GEMM:  C[m, z, n] = sum_k A[m, z, k] * B[z, k, n]
//   A element (m,k,z) at A[z*Az + m*ldA + k]   (f16, k contiguous)
//   B element (k,n,z) at B[z*Bz + n*Bn + k]    (f16, k contiguous)
//   C element (m,n,z) at C[m*ldC + z*Cz + n]   (f32)
// REQUIRES: K % 32 == 0; ldA, Az, Bn, Bz multiples of 8 halfs (16B).
// One 16x16 tile per wave; vector b128 loads, no per-element predication.
// A/B in contiguous k order (lane half h holds k = 16h..16h+15); 32-bit C 16x16 per CDNA5 ISA 7.12.2.
// ---------------------------------------------------------------------------
__global__ void gemm_wmma_f16_kernel(const _Float16* __restrict__ A, int ldA, int Az,
                                     const _Float16* __restrict__ Bm, long Bz, int Bn,
                                     float* __restrict__ C, int ldC, int Cz,
                                     int M, int N, int K)
{
  const int z    = blockIdx.z;
  const int lane = threadIdx.x & 31;
  const int wave = threadIdx.x >> 5;
  const int m0   = blockIdx.x * 16;
  const int n0   = (blockIdx.y * (blockDim.x >> 5) + wave) * 16;
  if (n0 >= N) return;   // wave-uniform: EXEC stays all-ones for WMMA

  const int arow = m0 + (lane & 15);
  const int akb  = (lane >> 4) * 16;   // A: lanes 16-31 hold the +16 K block
  const int bcol = n0 + (lane & 15);
  const int bkb  = (lane >> 4) * 16;   // B: lanes 16-31 hold the +16 K block

  const _Float16* ap = A  + (long)z * Az + (long)arow * ldA + akb;
  const _Float16* bp = Bm + (long)z * Bz + (long)bcol * Bn + bkb;

  v8f acc = {};
#pragma unroll 2
  for (int k0 = 0; k0 < K; k0 += 32) {
    v8h a0 = *(const v8h*)(ap + k0);        // K = k0+akb   .. +7
    v8h a1 = *(const v8h*)(ap + k0 + 8);    // K = k0+akb+8 .. +15
    v16h a = __builtin_shufflevector(a0, a1, 0, 1, 2, 3, 4, 5, 6, 7,
                                             8, 9, 10, 11, 12, 13, 14, 15);
    v16h b = *(const v16h*)(bp + k0);       // K = k0+bkb .. +15
    acc = __builtin_amdgcn_wmma_f32_16x16x32_f16(false, a, false, b, (short)0, acc,
                                                 false, false);
  }

  const int cn = n0 + (lane & 15);
  const int cm = m0 + (lane >> 4) * 8;      // VGPR g: M = g (lanes<16) / 8+g
#pragma unroll
  for (int g = 0; g < 8; ++g)
    C[(long)(cm + g) * ldC + (long)z * Cz + cn] = acc[g];
}

// ---------------------------------------------------------------------------
// f32 -> f16 convert with optional per-column bias / exact GELU / scale.
// ---------------------------------------------------------------------------
__global__ void cvt_f16_kernel(const float* __restrict__ x, const float* __restrict__ bias,
                               int bmod, float scale, int do_gelu,
                               _Float16* __restrict__ y, long n)
{
  long i      = (long)blockIdx.x * blockDim.x + threadIdx.x;
  long stride = (long)gridDim.x * blockDim.x;
  for (; i < n; i += stride) {
    float v = x[i];
    if (bias) v += bias[i % bmod];
    if (do_gelu) v = gelu_exact(v);
    y[i] = (_Float16)(v * scale);
  }
}

// q_pack[b, h, k(96)] = k<80 ? f16((qpre[b, h*80+k] + bq[h*80+k]) / sqrt(80)) : 0
__global__ void build_q_kernel(const float* __restrict__ qpre, const float* __restrict__ bq,
                               _Float16* __restrict__ qp)
{
  const float rs = 0.11180339887498949f;
  long i = (long)blockIdx.x * blockDim.x + threadIdx.x;
  long n = (long)B_ * H_ * DHP_;
  long stride = (long)gridDim.x * blockDim.x;
  for (; i < n; i += stride) {
    int  k  = (int)(i % DHP_);
    long bh = i / DHP_;
    int  h  = (int)(bh % H_);
    long b  = bh / H_;
    float v = 0.f;
    if (k < DH_) v = (qpre[b * D_ + h * DH_ + k] + bq[h * DH_ + k]) * rs;
    qp[i] = (_Float16)v;
  }
}

// wkT[h, n(640), k(96)] = k<80 ? f16(attn_in_w[(D + h*80 + k)*D + n]) : 0
__global__ void build_wkT_kernel(const float* __restrict__ aiw, _Float16* __restrict__ wkT)
{
  long i = (long)blockIdx.x * blockDim.x + threadIdx.x;
  long n = (long)H_ * D_ * DHP_;
  long stride = (long)gridDim.x * blockDim.x;
  for (; i < n; i += stride) {
    int  k  = (int)(i % DHP_);
    long hn = i / DHP_;
    int  c  = (int)(hn % D_);
    int  h  = (int)(hn / D_);
    float v = 0.f;
    if (k < DH_) v = aiw[(long)(D_ + h * DH_ + k) * D_ + c];
    wkT[i] = (_Float16)v;
  }
}

// ---------------------------------------------------------------------------
// Gather rows at edit_pos: xbg (f32 + f16) and (f_ed - f_bg) as f16.
// ---------------------------------------------------------------------------
__global__ void gather_kernel(const float* __restrict__ fb, const float* __restrict__ fe,
                              const int* __restrict__ pos,
                              float* __restrict__ xbg, _Float16* __restrict__ xbg16,
                              _Float16* __restrict__ xd16)
{
  int b = blockIdx.x;
  int p = pos[b];
  const float* rb = fb + ((long)b * S_ + p) * D_;
  const float* re = fe + ((long)b * S_ + p) * D_;
  for (int d = threadIdx.x; d < D_; d += blockDim.x) {
    float vb = rb[d];
    xbg  [(long)b * D_ + d] = vb;
    xbg16[(long)b * D_ + d] = (_Float16)vb;
    xd16 [(long)b * D_ + d] = (_Float16)(re[d] - vb);
  }
}

// ---------------------------------------------------------------------------
// Block reduction of (sum, sumsq); sh must hold >= 16 floats.
// ---------------------------------------------------------------------------
__device__ __forceinline__ void block_reduce_2(float& s, float& s2, float* sh)
{
  for (int o = 16; o; o >>= 1) { s += __shfl_xor(s, o); s2 += __shfl_xor(s2, o); }
  int w = threadIdx.x >> 5, l = threadIdx.x & 31;
  int nw = blockDim.x >> 5;
  if (l == 0) { sh[w] = s; sh[8 + w] = s2; }
  __syncthreads();
  if (threadIdx.x == 0) {
    float a = 0.f, b = 0.f;
    for (int i = 0; i < nw; ++i) { a += sh[i]; b += sh[8 + i]; }
    sh[0] = a; sh[8] = b;
  }
  __syncthreads();
  s = sh[0]; s2 = sh[8];
}

// out = LN( maybe_gelu(x + bias) ) * g + bb        (row per block)
__global__ void bias_act_ln_kernel(const float* __restrict__ x, const float* __restrict__ bias,
                                   const float* __restrict__ g, const float* __restrict__ bb,
                                   float* __restrict__ out, int N, int do_gelu)
{
  __shared__ float buf[D_];
  __shared__ float red[16];
  long r = blockIdx.x;
  float s = 0.f, s2 = 0.f;
  for (int d = threadIdx.x; d < N; d += blockDim.x) {
    float v = x[r * N + d] + bias[d];
    if (do_gelu) v = gelu_exact(v);
    buf[d] = v; s += v; s2 += v * v;
  }
  block_reduce_2(s, s2, red);
  float mean = s / N, var = s2 / N - mean * mean;
  float inv = rsqrtf(var + 1e-5f);
  for (int d = threadIdx.x; d < N; d += blockDim.x)
    out[r * N + d] = (buf[d] - mean) * inv * g[d] + bb[d];
}

// fused[:, :D] = f16( LN(ctxp + aob + xbg)*g+bb + local_delta )
__global__ void combine_ln_kernel(const float* __restrict__ ctxp, const float* __restrict__ bias,
                                  const float* __restrict__ xbg, const float* __restrict__ g,
                                  const float* __restrict__ bb, const float* __restrict__ ld,
                                  _Float16* __restrict__ fused)
{
  __shared__ float buf[D_];
  __shared__ float red[16];
  long r = blockIdx.x;
  float s = 0.f, s2 = 0.f;
  for (int d = threadIdx.x; d < D_; d += blockDim.x) {
    float v = ctxp[r * D_ + d] + bias[d] + xbg[r * D_ + d];
    buf[d] = v; s += v; s2 += v * v;
  }
  block_reduce_2(s, s2, red);
  float mean = s / D_, var = s2 / D_ - mean * mean;
  float inv = rsqrtf(var + 1e-5f);
  for (int d = threadIdx.x; d < D_; d += blockDim.x) {
    float v = (buf[d] - mean) * inv * g[d] + bb[d] + ld[r * D_ + d];
    fused[r * FUSE_ + d] = (_Float16)v;
  }
}

// ---------------------------------------------------------------------------
// Fused single-position attention, one block per batch element.
// scores[h,s] = qk[b,h,:] . f[b,s,:]   (scale already folded into qk)
// softmax over s (with mask), then fbar[b,h,:] = sum_s w[h,s] f[b,s,:] -> f16
// ---------------------------------------------------------------------------
__global__ void attn_kernel(const float* __restrict__ f, const float* __restrict__ qk,
                            const unsigned char* __restrict__ mask,
                            _Float16* __restrict__ fbar16)
{
  __shared__ float s_qk[H_ * D_];   // 20 KB
  __shared__ float s_sc[H_ * S_];   // 16 KB
  int b = blockIdx.x;
  int tid = threadIdx.x;            // 256 threads = 8 waves
  int lane = tid & 31, wave = tid >> 5;

  for (int i = tid; i < H_ * D_; i += blockDim.x) s_qk[i] = qk[(long)b * H_ * D_ + i];
  __syncthreads();

  // phase 1: scores (one s per wave iteration; lanes split D)
  for (int s = wave; s < S_; s += 8) {
    const float* frow = f + ((long)b * S_ + s) * D_;
    float fr[20];
#pragma unroll
    for (int i = 0; i < 20; ++i) fr[i] = frow[lane + 32 * i];
    bool ok = mask[(long)b * S_ + s] != 0;
#pragma unroll
    for (int h = 0; h < H_; ++h) {
      float p = 0.f;
#pragma unroll
      for (int i = 0; i < 20; ++i) p += fr[i] * s_qk[h * D_ + lane + 32 * i];
      for (int o = 16; o; o >>= 1) p += __shfl_xor(p, o);
      if (lane == 0) s_sc[h * S_ + s] = ok ? p : -3.0e38f;
    }
  }
  __syncthreads();

  // phase 2: softmax per head (wave w handles head w)
  {
    int h = wave;
    float mx = -3.4e38f;
    for (int s = lane; s < S_; s += 32) mx = fmaxf(mx, s_sc[h * S_ + s]);
    for (int o = 16; o; o >>= 1) mx = fmaxf(mx, __shfl_xor(mx, o));
    float sum = 0.f;
    for (int s = lane; s < S_; s += 32) {
      float e = __expf(s_sc[h * S_ + s] - mx);
      s_sc[h * S_ + s] = e; sum += e;
    }
    for (int o = 16; o; o >>= 1) sum += __shfl_xor(sum, o);
    float inv = 1.0f / sum;
    for (int s = lane; s < S_; s += 32) s_sc[h * S_ + s] *= inv;
  }
  __syncthreads();

  // phase 3: fbar accumulation; thread owns d = tid (+256, +512 if < 640)
  float acc[3][H_];
#pragma unroll
  for (int j = 0; j < 3; ++j)
#pragma unroll
    for (int h = 0; h < H_; ++h) acc[j][h] = 0.f;
  int nd = (tid < D_ - 512) ? 3 : 2;
  for (int s = 0; s < S_; ++s) {
    const float* frow = f + ((long)b * S_ + s) * D_;
    if (s + 1 < S_) __builtin_prefetch(f + ((long)b * S_ + s + 1) * D_ + tid, 0, 1);
    float wv[H_];
#pragma unroll
    for (int h = 0; h < H_; ++h) wv[h] = s_sc[h * S_ + s];
    for (int j = 0; j < nd; ++j) {
      float fv = frow[tid + 256 * j];
#pragma unroll
      for (int h = 0; h < H_; ++h) acc[j][h] += wv[h] * fv;
    }
  }
  for (int j = 0; j < nd; ++j) {
    int d = tid + 256 * j;
#pragma unroll
    for (int h = 0; h < H_; ++h)
      fbar16[((long)b * H_ + h) * D_ + d] = (_Float16)acc[j][h];
  }
}

// ---------------------------------------------------------------------------
// Small features: struct MLP (7->64->64 + LN), concordance (5->32 + LN),
// flank embedding lookup. One block (64 threads) per batch element.
// Writes fused[:, 640:768] as f16.
// ---------------------------------------------------------------------------
__global__ void small_feat_kernel(const float* __restrict__ sd, const float* __restrict__ cf,
                                  const float* __restrict__ embf, const int* __restrict__ fc,
                                  const float* __restrict__ sw1, const float* __restrict__ sb1,
                                  const float* __restrict__ sw2, const float* __restrict__ sb2,
                                  const float* __restrict__ sg,  const float* __restrict__ sbb,
                                  const float* __restrict__ cw,  const float* __restrict__ cb,
                                  const float* __restrict__ cg,  const float* __restrict__ cbb,
                                  _Float16* __restrict__ fused)
{
  int b = blockIdx.x, t = threadIdx.x;          // 64 threads = 2 waves
  __shared__ float sin7[7], cin5[5], h1[64], wsum[2][2];
  if (t < 7)            sin7[t]     = sd[b * 7 + t];
  if (t >= 8 && t < 13) cin5[t - 8] = cf[b * 5 + (t - 8)];
  __syncthreads();

  // struct layer 1 + gelu
  float a = sb1[t];
#pragma unroll
  for (int j = 0; j < 7; ++j) a += sw1[t * 7 + j] * sin7[j];
  h1[t] = gelu_exact(a);
  __syncthreads();

  // struct layer 2 + LN over 64
  float h2 = sb2[t];
  for (int j = 0; j < 64; ++j) h2 += sw2[t * 64 + j] * h1[j];
  float s = h2, s2 = h2 * h2;
  for (int o = 16; o; o >>= 1) { s += __shfl_xor(s, o); s2 += __shfl_xor(s2, o); }
  int w = t >> 5, l = t & 31;
  if (!l) { wsum[w][0] = s; wsum[w][1] = s2; }
  __syncthreads();
  float S1 = wsum[0][0] + wsum[1][0], S2 = wsum[0][1] + wsum[1][1];
  float m = S1 / 64.f, var = S2 / 64.f - m * m;
  fused[(long)b * FUSE_ + 672 + t] =
      (_Float16)((h2 - m) * rsqrtf(var + 1e-5f) * sg[t] + sbb[t]);
  __syncthreads();

  if (t < 32) {           // wave 0: concordance 5->32 + gelu + LN over 32
    float c = cb[t];
#pragma unroll
    for (int j = 0; j < 5; ++j) c += cw[t * 5 + j] * cin5[j];
    c = gelu_exact(c);
    float cs = c, cs2 = c * c;
    for (int o = 16; o; o >>= 1) { cs += __shfl_xor(cs, o); cs2 += __shfl_xor(cs2, o); }
    float cm = cs / 32.f, cv = cs2 / 32.f - cm * cm;
    fused[(long)b * FUSE_ + 736 + t] =
        (_Float16)((c - cm) * rsqrtf(cv + 1e-5f) * cg[t] + cbb[t]);
  } else {                // wave 1: flank embedding
    int i = t - 32;
    fused[(long)b * FUSE_ + 640 + i] = (_Float16)(embf[fc[b] * 32 + i]);
  }
}

// ===========================================================================
extern "C" void kernel_launch(void* const* d_in, const int* in_sizes, int n_in,
                              void* d_out, int out_size, void* d_ws, size_t ws_size,
                              hipStream_t stream)
{
  (void)in_sizes; (void)n_in; (void)out_size; (void)ws_size;

  const float* f_bg  = (const float*)d_in[0];
  const float* f_ed  = (const float*)d_in[1];
  const float* sdel  = (const float*)d_in[2];
  const float* conc  = (const float*)d_in[3];
  const float* embf  = (const float*)d_in[4];
  const float* ld_w  = (const float*)d_in[5];
  const float* ld_b  = (const float*)d_in[6];
  const float* ld_g  = (const float*)d_in[7];
  const float* ld_bb = (const float*)d_in[8];
  const float* aiw   = (const float*)d_in[9];   // (3D, D)
  const float* aib   = (const float*)d_in[10];  // (3D,)
  const float* aow   = (const float*)d_in[11];
  const float* aob   = (const float*)d_in[12];
  const float* cn_g  = (const float*)d_in[13];
  const float* cn_b  = (const float*)d_in[14];
  const float* sw1   = (const float*)d_in[15];
  const float* sb1   = (const float*)d_in[16];
  const float* sw2   = (const float*)d_in[17];
  const float* sb2   = (const float*)d_in[18];
  const float* sg    = (const float*)d_in[19];
  const float* sbb   = (const float*)d_in[20];
  const float* cw    = (const float*)d_in[21];
  const float* cb    = (const float*)d_in[22];
  const float* cg    = (const float*)d_in[23];
  const float* cbb   = (const float*)d_in[24];
  const float* fw1   = (const float*)d_in[25];  // (512, 768)
  const float* fb1   = (const float*)d_in[26];
  const float* fw2   = (const float*)d_in[27];  // (256, 512)
  const float* fb2   = (const float*)d_in[28];
  const float* fg    = (const float*)d_in[29];
  const float* fbb   = (const float*)d_in[30];
  const int*   epos  = (const int*)d_in[31];
  const int*   fc    = (const int*)d_in[32];
  const unsigned char* mask = (const unsigned char*)d_in[33];
  float* out = (float*)d_out;

  // ---- workspace carve-out (512B-aligned slices) ----
  char* p = (char*)d_ws;
  auto carve = [&](size_t bytes) { void* r = (void*)p; p += (bytes + 511) & ~(size_t)511; return r; };
  _Float16* ld_w16  = (_Float16*)carve((size_t)D_ * D_ * 2);
  _Float16* attn16  = (_Float16*)carve((size_t)3 * D_ * D_ * 2);   // wq | wk | wv
  _Float16* aow16   = (_Float16*)carve((size_t)D_ * D_ * 2);
  _Float16* fw116   = (_Float16*)carve((size_t)512 * FUSE_ * 2);
  _Float16* fw216   = (_Float16*)carve((size_t)DE_ * 512 * 2);
  _Float16* wkT16   = (_Float16*)carve((size_t)H_ * D_ * DHP_ * 2); // (H, n, kpad)
  float*    xbg     = (float*)   carve((size_t)B_ * D_ * 4);
  _Float16* xbg16   = (_Float16*)carve((size_t)B_ * D_ * 2);
  _Float16* xd16    = (_Float16*)carve((size_t)B_ * D_ * 2);
  float*    t0      = (float*)   carve((size_t)B_ * D_ * 4);
  float*    ld_out  = (float*)   carve((size_t)B_ * D_ * 4);
  _Float16* qp16    = (_Float16*)carve((size_t)B_ * H_ * DHP_ * 2); // (B, H, kpad)
  float*    qkbuf   = (float*)   carve((size_t)B_ * H_ * D_ * 4);
  _Float16* fbar16  = (_Float16*)carve((size_t)B_ * H_ * D_ * 2);
  float*    ctx     = (float*)   carve((size_t)B_ * D_ * 4);
  _Float16* ctx16   = (_Float16*)carve((size_t)B_ * D_ * 2);
  _Float16* fused16 = (_Float16*)carve((size_t)B_ * FUSE_ * 2);
  float*    t1      = (float*)   carve((size_t)B_ * 512 * 4);
  _Float16* h116    = (_Float16*)carve((size_t)B_ * 512 * 2);
  float*    t2      = (float*)   carve((size_t)B_ * DE_ * 4);

  auto cvt = [&](const float* x, const float* bias, int bmod, float scale, int gelu,
                 _Float16* y, long n) {
    int blocks = (int)((n + 255) / 256); if (blocks > 2048) blocks = 2048;
    cvt_f16_kernel<<<blocks, 256, 0, stream>>>(x, bias, bmod, scale, gelu, y, n);
  };
  auto gemm = [&](const _Float16* A, int ldA, int Az, const _Float16* Bm, long Bz, int Bn,
                  float* C, int ldC, int Cz, int M, int N, int K, int Z) {
    dim3 grid(M / 16, (((N + 15) / 16) + 3) / 4, Z);
    gemm_wmma_f16_kernel<<<grid, dim3(128), 0, stream>>>(A, ldA, Az, Bm, Bz, Bn,
                                                         C, ldC, Cz, M, N, K);
  };

  // 0) weight conversions to f16 (+ padded/transposed Wk for the qk GEMM)
  cvt(ld_w, nullptr, 1, 1.f, 0, ld_w16, (long)D_ * D_);
  cvt(aiw,  nullptr, 1, 1.f, 0, attn16, (long)3 * D_ * D_);
  cvt(aow,  nullptr, 1, 1.f, 0, aow16,  (long)D_ * D_);
  cvt(fw1,  nullptr, 1, 1.f, 0, fw116,  (long)512 * FUSE_);
  cvt(fw2,  nullptr, 1, 1.f, 0, fw216,  (long)DE_ * 512);
  build_wkT_kernel<<<1024, 256, 0, stream>>>(aiw, wkT16);

  // 1) gather rows at edit_pos
  gather_kernel<<<B_, 256, 0, stream>>>(f_bg, f_ed, epos, xbg, xbg16, xd16);

  // 2) local_delta = LN(GELU(xdelta @ ld_w^T + ld_b))
  gemm(xd16, D_, 0, ld_w16, 0, D_, t0, D_, 0, B_, D_, D_, 1);
  bias_act_ln_kernel<<<B_, 256, 0, stream>>>(t0, ld_b, ld_g, ld_bb, ld_out, D_, 1);

  // 3) q = xbg @ Wq^T; then pack (+bq, /sqrt(DH), zero-pad K to 96)
  gemm(xbg16, D_, 0, attn16, 0, D_, t0, D_, 0, B_, D_, D_, 1);
  build_q_kernel<<<1024, 256, 0, stream>>>(t0, aib, qp16);

  // 4) qk[b,h,:] = Wk_h^T q_h[b]   (per-head GEMM, padded K=96)
  gemm(qp16, H_ * DHP_, DHP_, wkT16, (long)D_ * DHP_, DHP_,
       qkbuf, H_ * D_, D_, B_, D_, DHP_, H_);

  // 5) streaming attention: scores -> softmax -> fbar (f16)
  attn_kernel<<<B_, 256, 0, stream>>>(f_bg, qkbuf, mask, fbar16);

  // 6) ctx_h = fbar_h @ Wv_h^T   (bias bv added during f16 convert below)
  gemm(fbar16, H_ * D_, D_, attn16 + (long)2 * D_ * D_, (long)DH_ * D_, D_,
       ctx, D_, DH_, B_, DH_, D_, H_);
  cvt(ctx, aib + 2 * D_, D_, 1.f, 0, ctx16, (long)B_ * D_);

  // 7) attention output proj + residual + LN(cn) + local_delta -> fused[:, :640]
  gemm(ctx16, D_, 0, aow16, 0, D_, t0, D_, 0, B_, D_, D_, 1);
  combine_ln_kernel<<<B_, 256, 0, stream>>>(t0, aob, xbg, cn_g, cn_b, ld_out, fused16);

  // 8) small features -> fused[:, 640:768]
  small_feat_kernel<<<B_, 64, 0, stream>>>(sdel, conc, embf, fc, sw1, sb1, sw2, sb2,
                                           sg, sbb, cw, cb, cg, cbb, fused16);

  // 9) fusion MLP: GELU(fused @ fu_w1^T + b1) @ fu_w2^T + b2, then LN -> out
  gemm(fused16, FUSE_, 0, fw116, 0, FUSE_, t1, 512, 0, B_, 512, FUSE_, 1);
  cvt(t1, fb1, 512, 1.f, 1, h116, (long)B_ * 512);
  gemm(h116, 512, 0, fw216, 0, 512, t2, DE_, 0, B_, DE_, 512, 1);
  bias_act_ln_kernel<<<B_, 256, 0, stream>>>(t2, fb2, fg, fbb, out, DE_, 0);
}
